// GCNEncoder_35519379538031
// MI455X (gfx1250) — compile-verified, matrix operand order checked by reading, not run on hardware
//
#include <hip/hip_runtime.h>
#include <hip/hip_bf16.h>

typedef __bf16 bf16;
typedef __attribute__((ext_vector_type(8)))  __bf16 v8bf;
typedef __attribute__((ext_vector_type(16))) __bf16 v16bf;
typedef __attribute__((ext_vector_type(8)))  float  v8f;

#define FDIM 256          // F_IN == H == 256
#define BN_EPS 1e-5f

// ------------------------------------------------------------------ utilities
__global__ void fill_f32(float* __restrict__ p, float v, int n) {
    int i = blockIdx.x * blockDim.x + threadIdx.x;
    if (i < n) p[i] = v;
}

__global__ void f32_to_bf16(const float* __restrict__ x, bf16* __restrict__ y, int n) {
    int i = blockIdx.x * blockDim.x + threadIdx.x;
    if (i < n) y[i] = (bf16)x[i];
}

// Wt[n*256 + k] = (bf16) W[k*256 + n]   (so B-fragment loads are contiguous)
__global__ void w_transpose_bf16(const float* __restrict__ W, bf16* __restrict__ Wt) {
    int i = blockIdx.x * blockDim.x + threadIdx.x;   // 0 .. 65535
    int n = i >> 8, k = i & 255;
    Wt[i] = (bf16)W[k * FDIM + n];
}

__global__ void deg_count(const int* __restrict__ dst, float* __restrict__ deg, int E) {
    int e = blockIdx.x * blockDim.x + threadIdx.x;
    if (e < E) atomicAdd(&deg[dst[e]], 1.0f);
}

__global__ void rsqrt_inplace(float* __restrict__ d, int n) {
    int i = blockIdx.x * blockDim.x + threadIdx.x;
    if (i < n) d[i] = rsqrtf(d[i]);
}

// --------------------------------------------------------------- bf16 WMMA GEMM
// C[M,256] = A[M,256] @ W[256,256], A bf16 row-major, Bt = W^T bf16 ([n][k]).
// One 16x16 tile per wave, 8 waves per block, K unrolled 8x32.
__global__ __launch_bounds__(256)
void gemm_bf16_wmma(const bf16* __restrict__ A, const bf16* __restrict__ Bt,
                    float* __restrict__ C, int M) {
    const int lane = threadIdx.x & 31;
    const int wave = threadIdx.x >> 5;
    const int tiles = (M >> 4) * (FDIM >> 4);
    int tile = blockIdx.x * 8 + wave;
    if (tile >= tiles) return;
    const int mT = tile >> 4;           // 256/16 = 16 column tiles
    const int nT = tile & 15;

    const int half  = lane >> 4;        // 0: lanes 0-15, 1: lanes 16-31
    const int kbase = half * 8;
    const int row = mT * 16 + (lane & 15);
    const int col = nT * 16 + (lane & 15);

    const bf16* aP = A  + (size_t)row * FDIM + kbase;
    const bf16* bP = Bt + (size_t)col * FDIM + kbase;

    v8f acc = {};
#pragma unroll
    for (int k0 = 0; k0 < FDIM; k0 += 32) {
        v8bf alo = *(const v8bf*)(aP + k0);        // K = k0+kb .. +7   (16B)
        v8bf ahi = *(const v8bf*)(aP + k0 + 16);   // K = k0+kb+16 .. +23
        v8bf blo = *(const v8bf*)(bP + k0);
        v8bf bhi = *(const v8bf*)(bP + k0 + 16);
        v16bf av = __builtin_shufflevector(alo, ahi, 0,1,2,3,4,5,6,7,8,9,10,11,12,13,14,15);
        v16bf bv = __builtin_shufflevector(blo, bhi, 0,1,2,3,4,5,6,7,8,9,10,11,12,13,14,15);
        acc = __builtin_amdgcn_wmma_f32_16x16x32_bf16(false, av, false, bv,
                                                      (short)0, acc, false, false);
    }
    // D layout: VGPR r -> row (mT*16 + half*8 + r), col fixed
    const int mbase = mT * 16 + half * 8;
#pragma unroll
    for (int r = 0; r < 8; r++)
        C[(size_t)(mbase + r) * FDIM + col] = acc[r];
}

// ------------------------------------------------- self-loop init + edge scatter
// agg[i][c] = bias[c] + h[i][c]*dinv[i]^2      (self loop)
__global__ void agg_init(const float* __restrict__ h, const float* __restrict__ dinv,
                         const float* __restrict__ bias, float* __restrict__ agg, int N) {
    int idx = blockIdx.x * blockDim.x + threadIdx.x;   // float4 index
    if (idx >= N * 64) return;
    int row = idx >> 6, c4 = idx & 63;
    float dn = dinv[row]; dn *= dn;
    float4 v = ((const float4*)h)[idx];
    float4 b = ((const float4*)bias)[c4];
    float4 o = { v.x * dn + b.x, v.y * dn + b.y, v.z * dn + b.z, v.w * dn + b.w };
    ((float4*)agg)[idx] = o;
}

// one wave per edge: agg[dst] += h[src] * dinv[src]*dinv[dst]
__global__ __launch_bounds__(256)
void edge_scatter(const int* __restrict__ src, const int* __restrict__ dst,
                  const float* __restrict__ h, const float* __restrict__ dinv,
                  float* __restrict__ agg, int E) {
    int e = blockIdx.x * 8 + (threadIdx.x >> 5);
    int lane = threadIdx.x & 31;
    if (e >= E) return;
    int s = src[e], d = dst[e];
    float norm = dinv[s] * dinv[d];
    const float4* hs = (const float4*)(h + (size_t)s * FDIM);
    float* od = agg + (size_t)d * FDIM;
#pragma unroll
    for (int i = 0; i < 2; i++) {
        int q = lane + i * 32;
        float4 v = hs[q];
        atomicAdd(od + q * 4 + 0, v.x * norm);
        atomicAdd(od + q * 4 + 1, v.y * norm);
        atomicAdd(od + q * 4 + 2, v.z * norm);
        atomicAdd(od + q * 4 + 3, v.w * norm);
    }
}

// -------------------------------------------------------------------- batchnorm
// sums[0..255] = sum, sums[256..511] = sum of squares
__global__ __launch_bounds__(256)
void bn_stats(const float* __restrict__ x, float* __restrict__ sums, int N) {
    int c = threadIdx.x;
    float s = 0.f, q = 0.f;
    for (int r = blockIdx.x; r < N; r += gridDim.x) {
        float v = x[(size_t)r * FDIM + c];
        s += v; q += v * v;
    }
    atomicAdd(&sums[c], s);
    atomicAdd(&sums[FDIM + c], q);
}

// y = relu((x-mu)*rsqrt(var+eps)*g + b), in place; optional bf16 copy for next GEMM
__global__ __launch_bounds__(256)
void bn_apply_relu(float* __restrict__ x, const float* __restrict__ sums,
                   const float* __restrict__ gamma, const float* __restrict__ beta,
                   bf16* __restrict__ xb, int writeB, int N) {
    int c = threadIdx.x;
    float invN = 1.0f / (float)N;
    float mu  = sums[c] * invN;
    float var = sums[FDIM + c] * invN - mu * mu;
    float sc  = gamma[c] * rsqrtf(var + BN_EPS);
    float sh  = beta[c] - mu * sc;
    for (int r = blockIdx.x; r < N; r += gridDim.x) {
        size_t i = (size_t)r * FDIM + c;
        float v = fmaxf(x[i] * sc + sh, 0.0f);
        x[i] = v;
        if (writeB) xb[i] = (bf16)v;
    }
}

// ---------------------------------------------------------------------- pooling
__global__ __launch_bounds__(256)
void pool_sum(const float* __restrict__ h, const int* __restrict__ batch,
              float* __restrict__ pool, float* __restrict__ cnt, int N) {
    int c = threadIdx.x;
    for (int r = blockIdx.x; r < N; r += gridDim.x) {
        int g = batch[r];
        atomicAdd(&pool[(size_t)g * FDIM + c], h[(size_t)r * FDIM + c]);
        if (c == 0) atomicAdd(&cnt[g], 1.0f);
    }
}

__global__ void pool_div(const float* __restrict__ pool, const float* __restrict__ cnt,
                         float* __restrict__ out, int G) {
    int i = blockIdx.x * blockDim.x + threadIdx.x;
    if (i >= G * FDIM) return;
    out[i] = pool[i] / fmaxf(cnt[i >> 8], 1.0f);
}

// ===========================================================================
extern "C" void kernel_launch(void* const* d_in, const int* in_sizes, int n_in,
                              void* d_out, int out_size, void* d_ws, size_t ws_size,
                              hipStream_t stream) {
    const float* x     = (const float*)d_in[0];
    const int*   ei    = (const int*)  d_in[1];
    const int*   batch = (const int*)  d_in[2];
    const float* W1    = (const float*)d_in[3];
    const float* b1    = (const float*)d_in[4];
    const float* g1    = (const float*)d_in[5];
    const float* be1   = (const float*)d_in[6];
    const float* W2    = (const float*)d_in[7];
    const float* b2    = (const float*)d_in[8];
    const float* g2    = (const float*)d_in[9];
    const float* be2   = (const float*)d_in[10];
    float* out = (float*)d_out;

    const int N = in_sizes[0] / FDIM;      // 40000
    const int E = in_sizes[1] / 2;         // 640000
    const int G = out_size / FDIM;         // 64
    const int* src = ei;
    const int* dst = ei + E;

    // ---- workspace carve (256B aligned)
    char* ws = (char*)d_ws;
    size_t off = 0;
    auto carve = [&](size_t bytes) { void* p = ws + off; off += (bytes + 255) & ~(size_t)255; return p; };
    bf16*  xb   = (bf16*) carve((size_t)N * FDIM * 2);   // GEMM input (bf16)
    bf16*  w1t  = (bf16*) carve(FDIM * FDIM * 2);
    bf16*  w2t  = (bf16*) carve(FDIM * FDIM * 2);
    float* h    = (float*)carve((size_t)N * FDIM * 4);   // GEMM output
    float* agg  = (float*)carve((size_t)N * FDIM * 4);   // aggregated / BN in place
    float* dinv = (float*)carve((size_t)N * 4);
    float* sums = (float*)carve(2 * FDIM * 4);
    float* pool = (float*)carve((size_t)G * FDIM * 4);
    float* cnt  = (float*)carve((size_t)G * 4);
    (void)ws_size; (void)n_in;

    const int T = 256;
    auto cdiv = [](int a, int b) { return (a + b - 1) / b; };
    const int gemmBlocks = cdiv((N >> 4) * (FDIM >> 4), 8);
    const int edgeBlocks = cdiv(E, 8);
    const int rowBlocks  = 512;            // grid-stride row loops

    // ---- degree / norm (self loop folded into init value 1.0)
    fill_f32<<<cdiv(N, T), T, 0, stream>>>(dinv, 1.0f, N);
    deg_count<<<cdiv(E, T), T, 0, stream>>>(dst, dinv, E);
    rsqrt_inplace<<<cdiv(N, T), T, 0, stream>>>(dinv, N);

    // ---- precision prep
    f32_to_bf16<<<cdiv(N * FDIM, T), T, 0, stream>>>(x, xb, N * FDIM);
    w_transpose_bf16<<<cdiv(FDIM * FDIM, T), T, 0, stream>>>(W1, w1t);
    w_transpose_bf16<<<cdiv(FDIM * FDIM, T), T, 0, stream>>>(W2, w2t);

    // ================= layer 1 =================
    gemm_bf16_wmma<<<gemmBlocks, T, 0, stream>>>(xb, w1t, h, N);
    agg_init<<<cdiv(N * 64, T), T, 0, stream>>>(h, dinv, b1, agg, N);
    edge_scatter<<<edgeBlocks, T, 0, stream>>>(src, dst, h, dinv, agg, E);
    fill_f32<<<cdiv(2 * FDIM, T), T, 0, stream>>>(sums, 0.0f, 2 * FDIM);
    bn_stats<<<rowBlocks, T, 0, stream>>>(agg, sums, N);
    bn_apply_relu<<<rowBlocks, T, 0, stream>>>(agg, sums, g1, be1, xb, 1, N);

    // ================= layer 2 =================
    gemm_bf16_wmma<<<gemmBlocks, T, 0, stream>>>(xb, w2t, h, N);
    agg_init<<<cdiv(N * 64, T), T, 0, stream>>>(h, dinv, b2, agg, N);
    edge_scatter<<<edgeBlocks, T, 0, stream>>>(src, dst, h, dinv, agg, E);
    fill_f32<<<cdiv(2 * FDIM, T), T, 0, stream>>>(sums, 0.0f, 2 * FDIM);
    bn_stats<<<rowBlocks, T, 0, stream>>>(agg, sums, N);
    bn_apply_relu<<<rowBlocks, T, 0, stream>>>(agg, sums, g2, be2, xb, 0, N);

    // ================= mean pool =================
    fill_f32<<<cdiv(G * FDIM + G, T), T, 0, stream>>>(pool, 0.0f, G * FDIM);
    fill_f32<<<1, T, 0, stream>>>(cnt, 0.0f, G);
    pool_sum<<<rowBlocks, T, 0, stream>>>(agg, batch, pool, cnt, N);
    pool_div<<<cdiv(G * FDIM, T), T, 0, stream>>>(pool, cnt, out, G);
}
